// MaskedAttentionFaceHead_85633057947834
// MI455X (gfx1250) — compile-verified, matrix operand order checked by reading, not run on hardware
//
#include <hip/hip_runtime.h>

// MaskedAttentionFaceHead for MI455X (gfx1250, wave32, WMMA).
// Structure exploit: decoder input is spatially constant => conv_transpose has 9
// distinct spatial outputs, conv2 has 25; the conv branch collapses to tiny
// (8x128)@(128x128) WMMA GEMMs. Dominant work: per-position neighbor softmax +
// attended(32736x128)@W1a(128x128) via v_wmma_f32_16x16x32_f16.
// This round: B matrices pre-swizzled to WMMA fragment order (contiguous 32B per
// lane -> _b128 loads), A buffers zero-padded to 16 rows (no divergent
// per-element predicated loads), GEMM loops unrolled.

typedef __attribute__((ext_vector_type(16))) _Float16 v16h;
typedef __attribute__((ext_vector_type(8)))  _Float16 v8h;
typedef __attribute__((ext_vector_type(8)))  float    v8f;

constexpr int HH = 93, WW = 44, HWSZ = HH * WW;   // 4092
constexpr int BB = 8, HID = 128, LAT = 1024, MAXN = 24;

__device__ __forceinline__ float gelu_dev(float x) {
  return 0.5f * x * (1.0f + erff(x * 0.70710678118654752f));
}

__device__ __forceinline__ int wmma_k(int t, int half) {
  return (t < 8) ? (half * 8 + t) : (16 + half * 8 + (t - 8));
}

__device__ __forceinline__ v16h load_a_frag(const _Float16* __restrict__ Arow,
                                            int kbase, int half) {
  v8h lo = *(const v8h*)(Arow + kbase + half * 8);
  v8h hi = *(const v8h*)(Arow + kbase + 16 + half * 8);
  return __builtin_shufflevector(lo, hi, 0, 1, 2, 3, 4, 5, 6, 7,
                                 8, 9, 10, 11, 12, 13, 14, 15);
}

__device__ __forceinline__ v16h load_b_frag(const _Float16* __restrict__ swz,
                                            int kt, int n, int half) {
  return *(const v16h*)(swz + ((size_t)(kt * 128 + n) * 32) + half * 16);
}

__global__ void k_zero(float* __restrict__ p, int n) {
  for (int i = blockIdx.x * blockDim.x + threadIdx.x; i < n; i += gridDim.x * blockDim.x)
    p[i] = 0.0f;
}

// dst[((kt*128+n)*32) + half*16 + t] = src[kt*32 + wmma_k(t,half)][n]
__global__ void k_swz(const float* __restrict__ s, _Float16* __restrict__ d,
                      int K, int nmat) {
  int total = nmat * K * 128;
  for (int i = blockIdx.x * blockDim.x + threadIdx.x; i < total;
       i += gridDim.x * blockDim.x) {
    int mi = i / (K * 128);
    int i2 = i - mi * (K * 128);
    int kt = i2 >> 12;
    int r  = i2 & 4095;
    int n  = r >> 5;
    int q  = r & 31;
    int half = q >> 4, t = q & 15;
    int k = kt * 32 + wmma_k(t, half);
    d[i] = (_Float16)s[(size_t)mi * K * 128 + (size_t)k * 128 + n];
  }
}

__global__ void k_pad_lat(const float* __restrict__ s, _Float16* __restrict__ d) {
  int i = blockIdx.x * blockDim.x + threadIdx.x;
  if (i >= 16 * LAT) return;
  int row = i >> 10;
  d[i] = (row < BB) ? (_Float16)s[i] : (_Float16)0.0f;
}

__global__ void k_latent_proj(const _Float16* __restrict__ lat16,
                              const _Float16* __restrict__ Wl_swz,
                              const float* __restrict__ bl,
                              _Float16* __restrict__ lc16) {
  int wave = threadIdx.x >> 5, lane = threadIdx.x & 31;
  int m = lane & 15, half = lane >> 4;
  int n = wave * 16 + (lane & 15);
  const _Float16* Arow = lat16 + (size_t)m * LAT;
  v8f c = {};
#pragma unroll 8
  for (int kt = 0; kt < LAT / 32; ++kt) {
    v16h a = load_a_frag(Arow, kt * 32, half);
    v16h b = load_b_frag(Wl_swz, kt, n, half);
    c = __builtin_amdgcn_wmma_f32_16x16x32_f16(false, a, false, b, (short)0, c, false, false);
  }
#pragma unroll
  for (int v = 0; v < 8; ++v) {
    int mm = v + 8 * half;
    lc16[mm * HID + n] = (mm < BB) ? (_Float16)gelu_dev(c[v] + bl[n]) : (_Float16)0.0f;
  }
}

__global__ void k_tap_gemm(const _Float16* __restrict__ lc16,
                           const _Float16* __restrict__ Wt_swz,
                           const _Float16* __restrict__ W1b_swz,
                           const float* __restrict__ b1,
                           float* __restrict__ T, float* __restrict__ precomp) {
  int z = blockIdx.x;
  const _Float16* Bs = (z < 9) ? (Wt_swz + (size_t)z * HID * HID) : W1b_swz;
  int wave = threadIdx.x >> 5, lane = threadIdx.x & 31;
  int m = lane & 15, half = lane >> 4;
  int n = wave * 16 + (lane & 15);
  const _Float16* Arow = lc16 + (size_t)m * HID;
  v8f c = {};
#pragma unroll
  for (int kt = 0; kt < 4; ++kt) {
    v16h a = load_a_frag(Arow, kt * 32, half);
    v16h b = load_b_frag(Bs, kt, n, half);
    c = __builtin_amdgcn_wmma_f32_16x16x32_f16(false, a, false, b, (short)0, c, false, false);
  }
#pragma unroll
  for (int v = 0; v < 8; ++v) {
    int mm = v + 8 * half;
    if (mm < BB) {
      if (z < 9) T[((size_t)z * BB + mm) * HID + n] = c[v];
      else       precomp[mm * HID + n] = c[v] + b1[n];
    }
  }
}

__global__ void k_v1(const float* __restrict__ T, const float* __restrict__ bt,
                     float* __restrict__ V1) {
  int i = blockIdx.x * blockDim.x + threadIdx.x;
  if (i >= 9 * BB * HID) return;
  int ch = i & 127, b = (i >> 7) & 7, cs = i >> 10;
  int r3 = cs / 3, c3 = cs % 3;
  float acc = bt[ch];
  for (int dh = -1; dh <= 1; ++dh) {
    if ((r3 == 0 && dh < 0) || (r3 == 2 && dh > 0)) continue;
    for (int dw = -1; dw <= 1; ++dw) {
      if ((c3 == 0 && dw < 0) || (c3 == 2 && dw > 0)) continue;
      int tap = (1 - dh) * 3 + (1 - dw);   // transpose conv: flipped kernel
      acc += T[((size_t)tap * BB + b) * HID + ch];
    }
  }
  V1[i] = acc;
}

__global__ void k_gn_stats(const float* __restrict__ V1, float* __restrict__ mu,
                           float* __restrict__ rstd) {
  int t = threadIdx.x;
  if (t >= BB * 8) return;
  int b = t >> 3, g = t & 7;
  const float rcnt[3] = {1.f, (float)(HH - 2), 1.f};
  const float ccnt[3] = {1.f, (float)(WW - 2), 1.f};
  float s = 0.f, s2 = 0.f;
  for (int cs = 0; cs < 9; ++cs) {
    float wgt = rcnt[cs / 3] * ccnt[cs % 3];
    for (int ch = 0; ch < 16; ++ch) {
      float v = V1[((size_t)cs * BB + b) * HID + g * 16 + ch];
      s += wgt * v; s2 += wgt * v * v;
    }
  }
  float N = 16.0f * HWSZ;
  float m = s / N, var = s2 / N - m * m;
  mu[t] = m; rstd[t] = rsqrtf(var + 1e-5f);
}

__global__ void k_v1norm(const float* __restrict__ V1, const float* __restrict__ mu,
                         const float* __restrict__ rstd, const float* __restrict__ g1,
                         const float* __restrict__ b1g, _Float16* __restrict__ V1n16) {
  int i = blockIdx.x * blockDim.x + threadIdx.x;
  if (i >= 9 * 16 * HID) return;
  int ch = i & 127, b = (i >> 7) & 15, cs = i >> 11;
  if (b >= BB) { V1n16[i] = (_Float16)0.0f; return; }
  int gi = b * 8 + (ch >> 4);
  float x = (V1[((size_t)cs * BB + b) * HID + ch] - mu[gi]) * rstd[gi] * g1[ch] + b1g[ch];
  V1n16[i] = (_Float16)gelu_dev(x);
}

__global__ void k_region25(const _Float16* __restrict__ V1n16,
                           const _Float16* __restrict__ Wc_swz,
                           const float* __restrict__ bc, float* __restrict__ feat) {
  int cs = blockIdx.x;
  int R = cs / 5, C = cs % 5;
  const int hrep[5] = {0, 1, 4, HH - 2, HH - 1};
  const int wrep[5] = {0, 1, 4, WW - 2, WW - 1};
  int wave = threadIdx.x >> 5, lane = threadIdx.x & 31;
  int m = lane & 15, half = lane >> 4;
  int n = wave * 16 + (lane & 15);
  v8f c = {};
  for (int dh = -1; dh <= 1; ++dh) {
    int hp = hrep[R] + dh; if (hp < 0 || hp >= HH) continue;
    int r3 = (hp == 0) ? 0 : ((hp == HH - 1) ? 2 : 1);
    for (int dw = -1; dw <= 1; ++dw) {
      int wp = wrep[C] + dw; if (wp < 0 || wp >= WW) continue;
      int c3 = (wp == 0) ? 0 : ((wp == WW - 1) ? 2 : 1);
      const _Float16* Arow = V1n16 + ((size_t)(r3 * 3 + c3) * 16 + m) * HID;
      const _Float16* Bs = Wc_swz + (size_t)((1 + dh) * 3 + (1 + dw)) * HID * HID;
#pragma unroll
      for (int kt = 0; kt < 4; ++kt) {
        v16h a = load_a_frag(Arow, kt * 32, half);
        v16h b = load_b_frag(Bs, kt, n, half);
        c = __builtin_amdgcn_wmma_f32_16x16x32_f16(false, a, false, b, (short)0, c, false, false);
      }
    }
  }
#pragma unroll
  for (int v = 0; v < 8; ++v) {
    int mm = v + 8 * half;
    if (mm < BB) feat[((size_t)mm * 25 + cs) * HID + n] = gelu_dev(c[v] + bc[n]);
  }
}

__global__ void k_logits25(const float* __restrict__ feat, const float* __restrict__ Wa,
                           const float* __restrict__ ba, float* __restrict__ logitT) {
  int t = blockIdx.x * blockDim.x + threadIdx.x;
  if (t >= BB * 25) return;
  const float4* f = (const float4*)(feat + (size_t)t * HID);
  const float4* w = (const float4*)Wa;
  float acc = ba[0];
#pragma unroll
  for (int j = 0; j < 32; ++j) {
    float4 a = f[j], bv = w[j];
    acc += a.x * bv.x + a.y * bv.y + a.z * bv.z + a.w * bv.w;
  }
  logitT[t] = acc;
}

__global__ void k_scan(const int* __restrict__ mask2d, int* __restrict__ tgt) {
  int b = blockIdx.x;
  const int* mrow = mask2d + (size_t)b * HWSZ;
  __shared__ int tot[256];
  int tid = threadIdx.x, base = tid * 16, s = 0;
  for (int j = 0; j < 16; ++j) { int p = base + j; s += (p < HWSZ && mrow[p] > 0); }
  tot[tid] = s; __syncthreads();
  for (int d = 1; d < 256; d <<= 1) {
    int v = (tid >= d) ? tot[tid - d] : 0;
    __syncthreads(); tot[tid] += v; __syncthreads();
  }
  int run = tot[tid] - s;
  for (int j = 0; j < 16; ++j) {
    int p = base + j;
    if (p < HWSZ) { run += (mrow[p] > 0); tgt[(size_t)b * HWSZ + p] = run - 1; }
  }
}

__global__ void __launch_bounds__(128)
k_attn(const int* __restrict__ mask2d, const int* __restrict__ nbrIdx,
       const int* __restrict__ nbrCnt, const float* __restrict__ feat,
       const float* __restrict__ logitT, const float* __restrict__ precomp,
       const _Float16* __restrict__ W1a_swz, const float* __restrict__ W2,
       const float* __restrict__ b2, const int* __restrict__ tgt,
       float* __restrict__ out) {
  int b = blockIdx.x >> 6;
  int pbase = (blockIdx.x & 63) * 64;
  int tid = threadIdx.x;
  __shared__ __align__(32) _Float16 Alds[64 * HID];
  __shared__ __align__(32) _Float16 Bswz[HID * HID];
  __shared__ __align__(32) _Float16 Hlds[64 * HID];
  __shared__ float W2s[HID * 2];
  {
    const uint4* src = (const uint4*)W1a_swz;
    uint4* dst = (uint4*)Bswz;
    for (int i = tid; i < HID * HID / 8; i += 128) dst[i] = src[i];
    for (int i = tid; i < HID * 2; i += 128) W2s[i] = W2[i];
  }
  int pl = tid >> 1, hs = tid & 1;
  int p = pbase + pl;
  float acc[64];
#pragma unroll
  for (int i = 0; i < 64; ++i) acc[i] = 0.0f;
  if (p < HWSZ) {
    int cnt = nbrCnt[p];
    float lg[MAXN]; int rg[MAXN];
    float mx = -1e30f;
    for (int nn = 0; nn < MAXN; ++nn) {
      int ni = nbrIdx[p * MAXN + nn];
      int hh = ni / WW, ww = ni - hh * WW;
      int r5 = (hh == 0) ? 0 : ((hh == 1) ? 1 : ((hh == HH - 2) ? 3 : ((hh == HH - 1) ? 4 : 2)));
      int c5 = (ww == 0) ? 0 : ((ww == 1) ? 1 : ((ww == WW - 2) ? 3 : ((ww == WW - 1) ? 4 : 2)));
      rg[nn] = r5 * 5 + c5;
      bool valid = (nn < cnt) && (mask2d[(size_t)b * HWSZ + ni] == 0);
      float l = valid ? logitT[b * 25 + rg[nn]] : -10000.0f;
      lg[nn] = l; mx = fmaxf(mx, l);
    }
    float ssum = 0.0f;
#pragma unroll
    for (int nn = 0; nn < MAXN; ++nn) { lg[nn] = __expf(lg[nn] - mx); ssum += lg[nn]; }
    float inv = 1.0f / ssum;
    for (int nn = 0; nn < MAXN; ++nn) {
      float wgt = lg[nn] * inv;
      if (wgt != 0.0f) {
        const float4* f4 = (const float4*)(feat + ((size_t)b * 25 + rg[nn]) * HID + hs * 64);
#pragma unroll
        for (int j = 0; j < 16; ++j) {
          float4 v = f4[j];
          acc[4 * j + 0] += wgt * v.x; acc[4 * j + 1] += wgt * v.y;
          acc[4 * j + 2] += wgt * v.z; acc[4 * j + 3] += wgt * v.w;
        }
      }
    }
  }
#pragma unroll
  for (int j = 0; j < 8; ++j) {
    v8h hv;
#pragma unroll
    for (int e = 0; e < 8; ++e) hv[e] = (_Float16)acc[8 * j + e];
    *(v8h*)(Alds + pl * HID + hs * 64 + 8 * j) = hv;
  }
  __syncthreads();
  int wave = tid >> 5, lane = tid & 31;
  int m = lane & 15, half = lane >> 4;
  const _Float16* Arow = Alds + (wave * 16 + m) * HID;
#pragma unroll
  for (int nt = 0; nt < 8; ++nt) {
    int n = nt * 16 + (lane & 15);
    v8f c = {};
#pragma unroll
    for (int kt = 0; kt < 4; ++kt) {
      v16h a = load_a_frag(Arow, kt * 32, half);
      v16h bb = load_b_frag(Bswz, kt, n, half);
      c = __builtin_amdgcn_wmma_f32_16x16x32_f16(false, a, false, bb, (short)0, c, false, false);
    }
    float pc = precomp[b * HID + n];
#pragma unroll
    for (int v = 0; v < 8; ++v) {
      int mm = v + 8 * half;
      Hlds[(wave * 16 + mm) * HID + n] = (_Float16)gelu_dev(c[v] + pc);
    }
  }
  __syncthreads();
  int pl3 = tid & 63, o = tid >> 6;
  int p3 = pbase + pl3;
  if (p3 < HWSZ && mask2d[(size_t)b * HWSZ + p3] > 0) {
    const _Float16* hrow = Hlds + pl3 * HID;
    float d = b2[o];
#pragma unroll
    for (int j = 0; j < 16; ++j) {
      v8h hv = *(const v8h*)(hrow + 8 * j);
#pragma unroll
      for (int e = 0; e < 8; ++e) d += (float)hv[e] * W2s[(8 * j + e) * 2 + o];
    }
    int slot = tgt[(size_t)b * HWSZ + p3];
    out[((size_t)b * HWSZ + slot) * 2 + o] = d;
    if (o == 0) {
      float* oidx = out + (size_t)2 * BB * HWSZ;
      oidx[((size_t)b * HWSZ + slot) * 2 + 0] = (float)(p3 / WW);
      oidx[((size_t)b * HWSZ + slot) * 2 + 1] = (float)(p3 % WW);
      float* oval = out + (size_t)4 * BB * HWSZ;
      oval[(size_t)b * HWSZ + slot] = 1.0f;
    }
  }
}

extern "C" void kernel_launch(void* const* d_in, const int* in_sizes, int n_in,
                              void* d_out, int out_size, void* d_ws, size_t ws_size,
                              hipStream_t stream) {
  const float* latent = (const float*)d_in[1];
  const int*   mask2d = (const int*)d_in[2];
  const int*   nbrIdx = (const int*)d_in[3];
  const int*   nbrCnt = (const int*)d_in[4];
  const float* Wl  = (const float*)d_in[5];
  const float* bl  = (const float*)d_in[6];
  const float* Wt  = (const float*)d_in[7];
  const float* bt  = (const float*)d_in[8];
  const float* g1  = (const float*)d_in[9];
  const float* b1g = (const float*)d_in[10];
  const float* Wc  = (const float*)d_in[11];
  const float* bc  = (const float*)d_in[12];
  const float* Wa  = (const float*)d_in[13];
  const float* ba  = (const float*)d_in[14];
  const float* W1  = (const float*)d_in[15];
  const float* b1  = (const float*)d_in[16];
  const float* W2  = (const float*)d_in[17];
  const float* b2  = (const float*)d_in[18];
  float* out = (float*)d_out;

  char* ws = (char*)d_ws;
  size_t off = 0;
  auto take = [&](size_t bytes) -> char* {
    char* p = ws + off; off = (off + bytes + 255) & ~(size_t)255; return p;
  };
  _Float16* lat16   = (_Float16*)take((size_t)16 * LAT * 2);
  _Float16* Wl_swz  = (_Float16*)take((size_t)LAT * HID * 2);
  _Float16* Wt_swz  = (_Float16*)take((size_t)9 * HID * HID * 2);
  _Float16* Wc_swz  = (_Float16*)take((size_t)9 * HID * HID * 2);
  _Float16* W1a_swz = (_Float16*)take((size_t)HID * HID * 2);
  _Float16* W1b_swz = (_Float16*)take((size_t)HID * HID * 2);
  _Float16* lc16    = (_Float16*)take((size_t)16 * HID * 2);
  float*    T       = (float*)take((size_t)9 * BB * HID * 4);
  float*    precomp = (float*)take((size_t)BB * HID * 4);
  float*    V1      = (float*)take((size_t)9 * BB * HID * 4);
  float*    gnMu    = (float*)take(64 * 4);
  float*    gnRstd  = (float*)take(64 * 4);
  _Float16* V1n16   = (_Float16*)take((size_t)9 * 16 * HID * 2);
  float*    feat    = (float*)take((size_t)BB * 25 * HID * 4);
  float*    logitT  = (float*)take((size_t)BB * 25 * 4);
  int*      tgt     = (int*)take((size_t)BB * HWSZ * 4);

  const int OUT_TOTAL = 5 * BB * HWSZ;
  k_zero<<<dim3(160), dim3(256), 0, stream>>>(out, OUT_TOTAL);

  k_pad_lat<<<dim3((16 * LAT + 255) / 256), dim3(256), 0, stream>>>(latent, lat16);
  k_swz<<<dim3((LAT * HID + 255) / 256), dim3(256), 0, stream>>>(Wl, Wl_swz, LAT, 1);
  k_swz<<<dim3((9 * HID * HID + 255) / 256), dim3(256), 0, stream>>>(Wt, Wt_swz, HID, 9);
  k_swz<<<dim3((9 * HID * HID + 255) / 256), dim3(256), 0, stream>>>(Wc, Wc_swz, HID, 9);
  k_swz<<<dim3((HID * HID + 255) / 256), dim3(256), 0, stream>>>(W1, W1a_swz, HID, 1);
  k_swz<<<dim3((HID * HID + 255) / 256), dim3(256), 0, stream>>>(W1 + HID * HID, W1b_swz, HID, 1);

  k_latent_proj<<<dim3(1), dim3(256), 0, stream>>>(lat16, Wl_swz, bl, lc16);
  k_tap_gemm<<<dim3(10), dim3(256), 0, stream>>>(lc16, Wt_swz, W1b_swz, b1, T, precomp);
  k_v1<<<dim3((9 * BB * HID + 255) / 256), dim3(256), 0, stream>>>(T, bt, V1);
  k_gn_stats<<<dim3(1), dim3(64), 0, stream>>>(V1, gnMu, gnRstd);
  k_v1norm<<<dim3((9 * 16 * HID + 255) / 256), dim3(256), 0, stream>>>(V1, gnMu, gnRstd, g1, b1g, V1n16);
  k_region25<<<dim3(25), dim3(256), 0, stream>>>(V1n16, Wc_swz, bc, feat);
  k_logits25<<<dim3(1), dim3(256), 0, stream>>>(feat, Wa, ba, logitT);
  k_scan<<<dim3(BB), dim3(256), 0, stream>>>(mask2d, tgt);
  k_attn<<<dim3(BB * 64), dim3(128), 0, stream>>>(mask2d, nbrIdx, nbrCnt, feat, logitT,
                                                  precomp, W1a_swz, W2, b2, tgt, out);
  (void)in_sizes; (void)n_in; (void)out_size; (void)ws_size;
}
